// DSSMModel_738734374914
// MI455X (gfx1250) — matrix operand order repaired and checked by reading, not compiled, not run on hardware
//
#include <hip/hip_runtime.h>

// ---------------------------------------------------------------------------
// DSSM: dual backward-GRU encoders + hc@hr^T + row softmax, on CDNA5 WMMA.
// B=1024, L=128, E=100 (pad->128), H=512. Fused per-step GEMM:
//   [1024 x 640] x [640 x 2048]  (4 gate blocks: r, z, i_n, h_n)
// bf16 operands, f32 accumulate via v_wmma_f32_16x16x32_bf16.
// ---------------------------------------------------------------------------

typedef __attribute__((ext_vector_type(16))) __bf16 v16bf;
typedef __attribute__((ext_vector_type(8)))  float  v8f;

#define BSZ   1024
#define LSEQ  128
#define EDIM  100
#define HDIM  512
#define KX    128          // padded E
#define KTOT  640          // KX + HDIM
#define KCH   20           // KTOT/32
#define NGATE 2048         // 4 * HDIM

__device__ __forceinline__ unsigned short f2bf(float f) {
    unsigned int u = __float_as_uint(f);
    unsigned int r = u + 0x7FFFu + ((u >> 16) & 1u);   // round-to-nearest-even
    return (unsigned short)(r >> 16);
}

// ---- pack combined gate weights into B-fragment layout: [kc<20][nc<128][512]
// B 32x16 bf16 frag: lane = n + 16*half; vgpr j<4 -> K=2j(+8*half); j>=4 -> K=16+2(j-4)(+8*half)
__global__ __launch_bounds__(256) void pack_weights(
    const float* __restrict__ Wih, const float* __restrict__ Whh,
    unsigned short* __restrict__ out)
{
    int i = blockIdx.x * 256 + threadIdx.x;          // < 20*128*512
    int pos  = i & 1;
    int j    = (i >> 1) & 7;
    int lane = (i >> 4) & 31;
    int nc   = (i >> 9) & 127;
    int kc   = i >> 16;
    int half = lane >> 4, n_in = lane & 15;
    int koff = ((j < 4) ? (2 * j + pos) : (16 + 2 * (j - 4) + pos)) + 8 * half;
    int k    = kc * 32 + koff;                        // 0..639
    int ncol = nc * 16 + n_in;                        // 0..2047
    int g    = ncol >> 9;                             // 0=r 1=z 2=i_n 3=h_n
    int c    = ncol & 511;
    float val = 0.0f;
    if (k < KX) {
        if (k < EDIM && g < 3) val = Wih[(size_t)(g * HDIM + c) * EDIM + k];
    } else {
        int kh = k - KX;
        if (g == 0)      val = Whh[(size_t)c * HDIM + kh];
        else if (g == 1) val = Whh[(size_t)(HDIM + c) * HDIM + kh];
        else if (g == 3) val = Whh[(size_t)(2 * HDIM + c) * HDIM + kh];
    }
    out[i] = f2bf(val);
}

__global__ void pack_bias(const float* __restrict__ bih,
                          const float* __restrict__ bhh,
                          float* __restrict__ out)
{
    int i = blockIdx.x * 256 + threadIdx.x;          // < 2048
    int g = i >> 9, c = i & 511;
    float v;
    if (g == 0)      v = bih[c] + bhh[c];
    else if (g == 1) v = bih[HDIM + c] + bhh[HDIM + c];
    else if (g == 2) v = bih[2 * HDIM + c];
    else             v = bhh[2 * HDIM + c];
    out[i] = v;
}

// ---- embedding gather (time-reversed) into A-fragment layout:
// Xfrag[t<128][mc<64][kc<4][512].  A 16x32 bf16 frag:
// lane = m + 16*half; vgpr j<4 -> K=2j(+8*half); j>=4 -> K=16+2(j-4)(+8*half)
__global__ __launch_bounds__(256) void gather_x(
    const int* __restrict__ tokens, const float* __restrict__ emb,
    unsigned short* __restrict__ out)
{
    int i = blockIdx.x * 256 + threadIdx.x;          // < 128*64*4*512
    int pos  = i & 1;
    int j    = (i >> 1) & 7;
    int lane = (i >> 4) & 31;
    int kc   = (i >> 9) & 3;
    int mc   = (i >> 11) & 63;
    int t    = i >> 17;
    int half = lane >> 4, m = lane & 15;
    int kk   = (j < 4) ? (2 * j + pos) : (16 + 2 * (j - 4) + pos);
    int koff = kk + 8 * half;
    int kx   = kc * 32 + koff;
    int b    = mc * 16 + m;
    int tok  = tokens[b * LSEQ + (LSEQ - 1 - t)];    // backward direction
    float v  = (kx < EDIM) ? emb[(size_t)tok * EDIM + kx] : 0.0f;
    out[i] = f2bf(v);
}

__global__ void zero_state(float* __restrict__ h, unsigned short* __restrict__ hf)
{
    int i = blockIdx.x * 256 + threadIdx.x;          // < 1024*512
    h[i]  = 0.0f;
    hf[i] = 0;
}

// ---- one GRU timestep: fused GEMM + gate epilogue.
// grid (32, 8): blockIdx.x = 16-col gate tile, blockIdx.y = 128-row tile.
// 8 waves/block; wave w owns rows [mt*128+w*16, +16), all 4 gate blocks.
__global__ __launch_bounds__(256) void gru_step(
    const unsigned short* __restrict__ Xfrag,
    const unsigned short* __restrict__ HfragIn,
    const unsigned short* __restrict__ Wcfrag,
    const float* __restrict__ bias,
    const float* __restrict__ hprev,
    float* __restrict__ hnext,
    unsigned short* __restrict__ HfragOut,
    int t)
{
    int lane = threadIdx.x & 31;
    int w    = threadIdx.x >> 5;
    int nt   = blockIdx.x;                 // 0..31
    int mt   = blockIdx.y;                 // 0..7
    int mc   = mt * 8 + w;                 // 0..63

    v8f acc[4] = {};
    const unsigned short* xbase = Xfrag + (((size_t)t * 64 + mc) * 4) * 512;
    const unsigned short* hbase = HfragIn + (size_t)mc * 16 * 512;

    for (int kc = 0; kc < KCH; ++kc) {
        const unsigned short* aptr = (kc < 4)
            ? (xbase + (size_t)kc * 512 + lane * 16)
            : (hbase + (size_t)(kc - 4) * 512 + lane * 16);
        v16bf a = *(const v16bf*)aptr;
        const unsigned short* bbase = Wcfrag + ((size_t)kc * 128 + nt) * 512;
#pragma unroll
        for (int g = 0; g < 4; ++g) {
            v16bf bb = *(const v16bf*)(bbase + (size_t)g * 32 * 512 + lane * 16);
            acc[g] = __builtin_amdgcn_wmma_f32_16x16x32_bf16(
                false, a, false, bb, (short)0, acc[g], false, false);
        }
    }

    // epilogue: lane holds r/z/i_n/h_n for the same (row, c) per acc vgpr.
    int c  = nt * 16 + (lane & 15);
    float br  = bias[c];
    float bz  = bias[HDIM + c];
    float bin = bias[2 * HDIM + c];
    float bhn = bias[3 * HDIM + c];
    int halfsel = lane >> 4;

    // h column -> next-step A-fragment position (K index = 128 + c)
    int khc   = c >> 5;
    int koffh = c & 31;
    int halfh = (koffh >> 3) & 1;
    int kkh   = koffh - 8 * halfh;
    int jh    = (kkh < 8) ? (kkh >> 1) : (4 + ((kkh - 16) >> 1));
    int posh  = kkh & 1;

#pragma unroll
    for (int v = 0; v < 8; ++v) {
        int m_in = v + 8 * halfsel;
        int row  = mc * 16 + m_in;
        float r  = 1.0f / (1.0f + __expf(-(acc[0][v] + br)));
        float z  = 1.0f / (1.0f + __expf(-(acc[1][v] + bz)));
        float nn = tanhf(acc[2][v] + bin + r * (acc[3][v] + bhn));
        float hp = hprev[(size_t)row * HDIM + c];
        float hv = (1.0f - z) * nn + z * hp;
        hnext[(size_t)row * HDIM + c] = hv;
        int lane2 = m_in + 16 * halfh;
        HfragOut[((size_t)mc * 16 + khc) * 512 + lane2 * 16 + jh * 2 + posh] = f2bf(hv);
    }
}

// ---- pack hr (f32 [1024][512]) into B-fragments for scores GEMM:
// hrB[kc<16][nc<64][512], B[k][n] = hr[n][k]
__global__ __launch_bounds__(256) void pack_hrB(
    const float* __restrict__ h, unsigned short* __restrict__ out)
{
    int i = blockIdx.x * 256 + threadIdx.x;          // < 16*64*512
    int pos  = i & 1;
    int j    = (i >> 1) & 7;
    int lane = (i >> 4) & 31;
    int nc   = (i >> 9) & 63;
    int kc   = i >> 15;
    int half = lane >> 4, n_in = lane & 15;
    int koff = ((j < 4) ? (2 * j + pos) : (16 + 2 * (j - 4) + pos)) + 8 * half;
    float v  = h[(size_t)(nc * 16 + n_in) * HDIM + kc * 32 + koff];
    out[i] = f2bf(v);
}

// ---- scores = hc @ hr^T, M=N=1024, K=512.  grid (16, 8), 8 waves/block,
// wave: 16 rows x 64 cols (4 wmma tiles).
__global__ __launch_bounds__(256) void scores_gemm(
    const unsigned short* __restrict__ hcA,
    const unsigned short* __restrict__ hrB,
    float* __restrict__ scores)
{
    int lane  = threadIdx.x & 31;
    int w     = threadIdx.x >> 5;
    int ntile = blockIdx.x;               // 0..15
    int mt    = blockIdx.y;               // 0..7
    int mc    = mt * 8 + w;               // 0..63

    v8f acc[4] = {};
    for (int kc = 0; kc < 16; ++kc) {
        v16bf a = *(const v16bf*)(hcA + ((size_t)mc * 16 + kc) * 512 + lane * 16);
#pragma unroll
        for (int j = 0; j < 4; ++j) {
            v16bf bb = *(const v16bf*)(hrB + ((size_t)kc * 64 + ntile * 4 + j) * 512 + lane * 16);
            acc[j] = __builtin_amdgcn_wmma_f32_16x16x32_bf16(
                false, a, false, bb, (short)0, acc[j], false, false);
        }
    }
    int nbase = ntile * 64 + (lane & 15);
    int half  = lane >> 4;
#pragma unroll
    for (int j = 0; j < 4; ++j)
#pragma unroll
        for (int v = 0; v < 8; ++v) {
            int row = mc * 16 + v + 8 * half;
            scores[(size_t)row * BSZ + nbase + j * 16] = acc[j][v];
        }
}

// ---- row softmax: one block per row, 4 elems per thread in registers.
__global__ __launch_bounds__(256) void softmax_rows(
    const float* __restrict__ scores, float* __restrict__ out)
{
    __shared__ float red[256];
    int row = blockIdx.x, tid = threadIdx.x;
    float v[4];
    float mx = -1e30f;
#pragma unroll
    for (int j = 0; j < 4; ++j) {
        v[j] = scores[(size_t)row * BSZ + tid + j * 256];
        mx = fmaxf(mx, v[j]);
    }
    red[tid] = mx; __syncthreads();
    for (int s = 128; s > 0; s >>= 1) {
        if (tid < s) red[tid] = fmaxf(red[tid], red[tid + s]);
        __syncthreads();
    }
    mx = red[0]; __syncthreads();
    float sum = 0.0f;
#pragma unroll
    for (int j = 0; j < 4; ++j) { v[j] = __expf(v[j] - mx); sum += v[j]; }
    red[tid] = sum; __syncthreads();
    for (int s = 128; s > 0; s >>= 1) {
        if (tid < s) red[tid] += red[tid + s];
        __syncthreads();
    }
    float inv = 1.0f / red[0];
#pragma unroll
    for (int j = 0; j < 4; ++j)
        out[(size_t)row * BSZ + tid + j * 256] = v[j] * inv;
}

extern "C" void kernel_launch(void* const* d_in, const int* in_sizes, int n_in,
                              void* d_out, int out_size, void* d_ws, size_t ws_size,
                              hipStream_t stream)
{
    (void)in_sizes; (void)n_in; (void)out_size; (void)ws_size;
    const int*   contexts = (const int*)d_in[0];
    const int*   replies  = (const int*)d_in[1];
    const float* embs[2]  = { (const float*)d_in[2], (const float*)d_in[7] };
    const float* Wihs[2]  = { (const float*)d_in[3], (const float*)d_in[8] };
    const float* Whhs[2]  = { (const float*)d_in[4], (const float*)d_in[9] };
    const float* bihs[2]  = { (const float*)d_in[5], (const float*)d_in[10] };
    const float* bhhs[2]  = { (const float*)d_in[6], (const float*)d_in[11] };
    const int*   toks[2]  = { contexts, replies };
    float* out = (float*)d_out;

    unsigned char* ws = (unsigned char*)d_ws;
    size_t off = 0;
    unsigned short* Xfrag  = (unsigned short*)(ws + off); off += (size_t)LSEQ * 64 * 4 * 512 * 2; // 32 MB
    unsigned short* Wcfrag = (unsigned short*)(ws + off); off += (size_t)KCH * 128 * 512 * 2;     // 2.5 MB
    float*          bias   = (float*)(ws + off);          off += (size_t)NGATE * 4;
    float*          hb[2];
    hb[0] = (float*)(ws + off); off += (size_t)BSZ * HDIM * 4;
    hb[1] = (float*)(ws + off); off += (size_t)BSZ * HDIM * 4;
    unsigned short* hf[2];
    hf[0] = (unsigned short*)(ws + off); off += (size_t)64 * 16 * 512 * 2;
    hf[1] = (unsigned short*)(ws + off); off += (size_t)64 * 16 * 512 * 2;
    unsigned short* hcA    = (unsigned short*)(ws + off); off += (size_t)64 * 16 * 512 * 2;
    unsigned short* hrB    = (unsigned short*)(ws + off); off += (size_t)16 * 64 * 512 * 2;
    float*          scores = (float*)(ws + off);          off += (size_t)BSZ * BSZ * 4;

    for (int gi = 0; gi < 2; ++gi) {
        pack_weights<<<(KCH * 128 * 512) / 256, 256, 0, stream>>>(Wihs[gi], Whhs[gi], Wcfrag);
        pack_bias<<<NGATE / 256, 256, 0, stream>>>(bihs[gi], bhhs[gi], bias);
        gather_x<<<((size_t)LSEQ * 64 * 4 * 512) / 256, 256, 0, stream>>>(toks[gi], embs[gi], Xfrag);
        zero_state<<<(BSZ * HDIM) / 256, 256, 0, stream>>>(hb[0], hf[0]);
        for (int t = 0; t < LSEQ; ++t) {
            gru_step<<<dim3(32, 8), 256, 0, stream>>>(
                Xfrag, hf[t & 1], Wcfrag, bias,
                hb[t & 1], hb[(t + 1) & 1], hf[(t + 1) & 1], t);
        }
        // after 128 (even) steps, final state sits in parity-0 buffers
        if (gi == 0) {
            hipMemcpyAsync(hcA, hf[0], (size_t)64 * 16 * 512 * 2,
                           hipMemcpyDeviceToDevice, stream);
        } else {
            pack_hrB<<<(16 * 64 * 512) / 256, 256, 0, stream>>>(hb[0], hrB);
        }
    }

    scores_gemm<<<dim3(16, 8), 256, 0, stream>>>(hcA, hrB, scores);
    softmax_rows<<<BSZ, 256, 0, stream>>>(scores, out);
}
